// AFNO2D_4449586119062
// MI455X (gfx1250) — matrix operand order repaired and checked by reading, not compiled, not run on hardware
//
#include <hip/hip_runtime.h>
#include <hip/hip_bf16.h>

// ---------------------------------------------------------------------------
// AFNO2D block for MI455X (gfx1250, wave32, WMMA).
// All matmul stages (pwconv GEMMs, DFT-as-GEMM FFTs, block-diag complex MLP)
// run through v_wmma_f32_16x16x32_bf16 with multi-accumulator wave tiles.
// The forward W-DFT stages its DFT matrix into LDS via the Tensor Data Mover
// (tensor_load_to_lds + s_wait_tensorcnt) when the builtin is available.
// Workspace requirement: ~1.58 GB.
// ---------------------------------------------------------------------------

typedef __attribute__((ext_vector_type(16))) __bf16 v16bf;
typedef __attribute__((ext_vector_type(8)))  float  v8f;

#define B_      32
#define H_      56
#define W_      56
#define WF_     29
#define CIN_    384
#define MED_    768
#define NB_     8
#define BS_     96
#define N1_     100352LL   // B*H*W spatial tokens
#define N2_     51968LL    // B*H*WF frequency tokens
#define LAMBDA_ 0.01f

#if defined(__has_builtin)
#if __has_builtin(__builtin_amdgcn_tensor_load_to_lds) && \
    __has_builtin(__builtin_amdgcn_s_wait_tensorcnt)
#define USE_TDM 1
#endif
#endif
#ifndef USE_TDM
#define USE_TDM 0
#endif

#if USE_TDM
typedef __attribute__((ext_vector_type(4))) unsigned int u32x4;
typedef __attribute__((ext_vector_type(4))) int          i32x4;
typedef __attribute__((ext_vector_type(8))) int          i32x8;
#endif

// ---------------- WMMA fragment helpers (A/B k = 16*(lane/16)+s; C/D per ISA 7.12.2) ------------

__device__ __forceinline__ v8f wmma_bf16(v16bf a, v16bf b, v8f c) {
  return __builtin_amdgcn_wmma_f32_16x16x32_bf16(false, a, false, b,
                                                 (short)0, c, false, false);
}

// A fragment: 16x32 (MxK) slice of fp32 row-major matrix -> bf16 in regs.
__device__ __forceinline__ v16bf load_a16x32(const float* __restrict__ A,
                                             long long row0, int lda, int k0,
                                             long long mvalid, int kvalid,
                                             float sgn) {
  const int lane = threadIdx.x & 31;
  const int m    = lane & 15;
  const int hi8  = (lane >> 4) << 3;
  const bool mok = (row0 + m) < mvalid;
  const float* rp = A + (long long)(row0 + m) * lda;
  v16bf a;
#pragma unroll
  for (int j = 0; j < 8; ++j) {
    const int kb = k0 + 2 * hi8 + 2 * j;
    float x0 = (mok && kb     < kvalid) ? rp[kb]     : 0.f;
    float x1 = (mok && kb + 1 < kvalid) ? rp[kb + 1] : 0.f;
    a[2 * j]     = (__bf16)(sgn * x0);
    a[2 * j + 1] = (__bf16)(sgn * x1);
  }
  return a;
}

// Same, but produce +A and -A fragments from one pass over memory.
__device__ __forceinline__ void load_a16x32_pm(const float* __restrict__ A,
                                               long long row0, int lda, int k0,
                                               int kvalid, v16bf& ap, v16bf& an) {
  const int lane = threadIdx.x & 31;
  const int m    = lane & 15;
  const int hi8  = (lane >> 4) << 3;
  const float* rp = A + (long long)(row0 + m) * lda;
#pragma unroll
  for (int j = 0; j < 8; ++j) {
    const int kb = k0 + 2 * hi8 + 2 * j;
    float x0 = (kb     < kvalid) ? rp[kb]     : 0.f;
    float x1 = (kb + 1 < kvalid) ? rp[kb + 1] : 0.f;
    ap[2 * j]     = (__bf16)x0;  an[2 * j]     = (__bf16)(-x0);
    ap[2 * j + 1] = (__bf16)x1;  an[2 * j + 1] = (__bf16)(-x1);
  }
}

// B fragment: 32x16 (KxN) slice of fp32 row-major matrix -> bf16 in regs.
__device__ __forceinline__ v16bf load_b32x16(const float* __restrict__ Bm,
                                             int k0, long long ldb, int n0,
                                             int kvalid, float sgn) {
  const int lane = threadIdx.x & 31;
  const int n    = n0 + (lane & 15);
  const int kh   = (lane >> 4) << 4;
  v16bf b;
#pragma unroll
  for (int e = 0; e < 16; ++e) {
    const int k = k0 + kh + e;
    float x = (k < kvalid) ? Bm[(long long)k * ldb + n] : 0.f;
    b[e] = (__bf16)(sgn * x);
  }
  return b;
}

// +B and -B fragments from one pass over memory.
__device__ __forceinline__ void load_b32x16_pm(const float* __restrict__ Bm,
                                               int k0, long long ldb, int n0,
                                               int kvalid, v16bf& bp, v16bf& bn) {
  const int lane = threadIdx.x & 31;
  const int n    = n0 + (lane & 15);
  const int kh   = (lane >> 4) << 4;
#pragma unroll
  for (int e = 0; e < 16; ++e) {
    const int k = k0 + kh + e;
    float x = (k < kvalid) ? Bm[(long long)k * ldb + n] : 0.f;
    bp[e] = (__bf16)x;
    bn[e] = (__bf16)(-x);
  }
}

// C/D store: VGPR j -> row (row0 + j + 8*(lane>=16)), col n0 + lane%16.
__device__ __forceinline__ void store_c(float* __restrict__ C, long long row0,
                                        long long ldc, int n0, long long mvalid,
                                        const v8f& acc) {
  const int lane = threadIdx.x & 31;
  const int n    = n0 + (lane & 15);
  const int mh   = (lane >> 4) << 3;
#pragma unroll
  for (int j = 0; j < 8; ++j) {
    const long long m = row0 + mh + j;
    if (m < mvalid) C[m * ldc + n] = acc[j];
  }
}

__device__ __forceinline__ long long wave_id() {
  return (long long)blockIdx.x * (blockDim.x >> 5) + (threadIdx.x >> 5);
}

// ---------------- DFT matrix generation (ortho-normalized) ------------------
//   [    0] fwdW cos 32x64, [ 2048] fwdW sin 32x64   (contiguous 16KB pair)
//   [ 4096] fwdH cos 64x64, [ 8192] fwdH -sin 64x64
//   [12288] invH cos 64x64, [16384] invH +sin 64x64
//   [20480] invW re  64x32, [22528] invW im  64x32
__global__ void init_dft(float* __restrict__ dft) {
  const float s  = 0.13363062095621219f;       // 1/sqrt(56)
  const float tp = 6.28318530717958647f;
  for (int i = threadIdx.x; i < 32 * 64; i += blockDim.x) {
    int wf = i >> 6, w = i & 63;
    float c = 0.f, sn = 0.f;
    if (wf < WF_ && w < W_) {
      float ang = tp * (float)(wf * w) / 56.f;
      c = __cosf(ang) * s; sn = -__sinf(ang) * s;
    }
    dft[i] = c; dft[2048 + i] = sn;
  }
  for (int i = threadIdx.x; i < 64 * 64; i += blockDim.x) {
    int a = i >> 6, b = i & 63;
    float c = 0.f, sn = 0.f;
    if (a < H_ && b < H_) {
      float ang = tp * (float)(a * b) / 56.f;
      c = __cosf(ang) * s; sn = __sinf(ang) * s;
    }
    dft[4096 + i]  = c;  dft[8192 + i]  = -sn;   // forward: e^{-i}
    dft[12288 + i] = c;  dft[16384 + i] = sn;    // inverse: e^{+i}
  }
  for (int i = threadIdx.x; i < 64 * 32; i += blockDim.x) {
    int w = i >> 5, wf = i & 31;
    float vr = 0.f, vi = 0.f;
    if (w < W_ && wf < WF_) {
      float cf = (wf == 0 || wf == WF_ - 1) ? 1.f : 2.f;  // Hermitian doubling
      float ang = tp * (float)(wf * w) / 56.f;
      vr =  cf * __cosf(ang) * s;
      vi = -cf * __sinf(ang) * s;
    }
    dft[20480 + i] = vr; dft[22528 + i] = vi;
  }
}

// ---------------- Stage 1: pwconv1 + StarReLU (wave tile 16x64) -------------
__global__ __launch_bounds__(256) void gemm_pw1_star(
    const float* __restrict__ X, const float* __restrict__ Wp,
    const float* __restrict__ sscale, const float* __restrict__ sbias,
    float* __restrict__ Hb) {
  const long long NTN = MED_ / 64;             // 12
  const long long wid = wave_id();
  if (wid >= (N1_ / 16) * NTN) return;
  const long long row0 = (wid / NTN) * 16;
  const int n0 = (int)(wid % NTN) * 64;
  v8f acc0 = {}, acc1 = {}, acc2 = {}, acc3 = {};
  for (int k0 = 0; k0 < CIN_; k0 += 32) {
    v16bf a = load_a16x32(X, row0, CIN_, k0, N1_, CIN_, 1.f);
    acc0 = wmma_bf16(a, load_b32x16(Wp, k0, MED_, n0,      CIN_, 1.f), acc0);
    acc1 = wmma_bf16(a, load_b32x16(Wp, k0, MED_, n0 + 16, CIN_, 1.f), acc1);
    acc2 = wmma_bf16(a, load_b32x16(Wp, k0, MED_, n0 + 32, CIN_, 1.f), acc2);
    acc3 = wmma_bf16(a, load_b32x16(Wp, k0, MED_, n0 + 48, CIN_, 1.f), acc3);
  }
  const float ss = sscale[0], sb = sbias[0];
  v8f* accs[4] = {&acc0, &acc1, &acc2, &acc3};
#pragma unroll
  for (int t = 0; t < 4; ++t) {
    v8f& a = *accs[t];
#pragma unroll
    for (int j = 0; j < 8; ++j) {
      float v = a[j] > 0.f ? a[j] : 0.f;
      a[j] = ss * v * v + sb;
    }
    store_c(Hb, row0, MED_, n0 + 16 * t, N1_, a);
  }
}

// ---------------- Stage 2: forward real DFT along W (wave tile 16x32) -------
// per (b,h): F[wf,c] = sum_w A[wf,w]*h[w,c]; A = [cos;sin] staged via TDM.
__global__ __launch_bounds__(256) void fwd_wdft(
    const float* __restrict__ Hb, const float* __restrict__ dft,
    float* __restrict__ Fr, float* __restrict__ Fi) {
#if USE_TDM
  __shared__ float sA[4096];                    // fwdW cos(2048) + sin(2048)
  if ((threadIdx.x >> 5) == 0) {
    const unsigned lds_off = (unsigned)(unsigned long long)(void*)&sA[0];
    const unsigned long long ga = (unsigned long long)(const void*)dft;
    u32x4 g0;
    g0[0] = 1u;                                 // count=1, user descriptor
    g0[1] = lds_off;                            // lds_addr (bytes)
    g0[2] = (unsigned)(ga & 0xffffffffull);     // global_addr[31:0]
    g0[3] = (unsigned)((ga >> 32) & 0x1ffffffull) | (2u << 30);  // type=image
    i32x8 g1;
    g1[0] = 0x00020000;                         // data_size = 4 bytes
    g1[1] = (int)((4096u & 0xffffu) << 16);     // tensor_dim0 = 4096
    g1[2] = (int)(1u << 16);                    // tensor_dim1 = 1
    g1[3] = (int)((4096u & 0xffffu) << 16);     // tile_dim0 = 4096 (1-D tile)
    g1[4] = 0;                                  // tile_dim1/2 = 0
    g1[5] = 4096;                               // tensor_dim0_stride
    g1[6] = 0;
    g1[7] = 0;
    i32x4 z4 = {0, 0, 0, 0};
#if __clang_major__ >= 23
    i32x8 z8 = {0, 0, 0, 0, 0, 0, 0, 0};
    __builtin_amdgcn_tensor_load_to_lds(g0, g1, z4, z4, z8, 0);
#else
    __builtin_amdgcn_tensor_load_to_lds(g0, g1, z4, z4, 0);
#endif
    __builtin_amdgcn_s_wait_tensorcnt(0);
  }
  __syncthreads();
  const float* Ad = &sA[0];
#else
  const float* Ad = dft;
#endif
  const long long total = (long long)B_ * H_ * 2 * 24;
  const long long wid = wave_id();
  if (wid >= total) return;
  const int slice = (int)(wid / 48);
  const int r = (int)(wid % 48);
  const int row0 = (r / 24) * 16, n0 = (r % 24) * 32;
  const float* Bp = Hb + (long long)slice * W_ * MED_;
  v8f ar0 = {}, ar1 = {}, ai0 = {}, ai1 = {};
#pragma unroll
  for (int k0 = 0; k0 < 64; k0 += 32) {
    v16bf ac = load_a16x32(Ad,        row0, 64, k0, 32, 64, 1.f);
    v16bf as = load_a16x32(Ad + 2048, row0, 64, k0, 32, 64, 1.f);
    v16bf b0 = load_b32x16(Bp, k0, MED_, n0,      W_, 1.f);
    v16bf b1 = load_b32x16(Bp, k0, MED_, n0 + 16, W_, 1.f);
    ar0 = wmma_bf16(ac, b0, ar0);  ar1 = wmma_bf16(ac, b1, ar1);
    ai0 = wmma_bf16(as, b0, ai0);  ai1 = wmma_bf16(as, b1, ai1);
  }
  const long long ob = (long long)slice * WF_ * MED_;
  store_c(Fr + ob, row0, MED_, n0,      WF_, ar0);
  store_c(Fr + ob, row0, MED_, n0 + 16, WF_, ar1);
  store_c(Fi + ob, row0, MED_, n0,      WF_, ai0);
  store_c(Fi + ob, row0, MED_, n0 + 16, WF_, ai1);
}

// ---------------- Stage 3/6: complex DFT along H (wave tile 16x32) ----------
// Gr = Ac*Xr - As*Xi, Gi = Ac*Xi + As*Xr
__global__ __launch_bounds__(256) void hdft_complex(
    const float* __restrict__ Xr, const float* __restrict__ Xi,
    const float* __restrict__ Ac, const float* __restrict__ As,
    float* __restrict__ Gr, float* __restrict__ Gi) {
  const long long total = (long long)B_ * WF_ * 4 * 24;
  const long long wid = wave_id();
  if (wid >= total) return;
  const int slice = (int)(wid / 96);
  const int r = (int)(wid % 96);
  const int row0 = (r / 24) * 16, n0 = (r % 24) * 32;
  const int bb = slice / WF_, wf = slice % WF_;
  const long long base = ((long long)bb * H_ * WF_ + wf) * MED_;
  const long long ldb = (long long)WF_ * MED_;
  v8f gr0 = {}, gr1 = {}, gi0 = {}, gi1 = {};
#pragma unroll
  for (int k0 = 0; k0 < 64; k0 += 32) {
    v16bf ac = load_a16x32(Ac, row0, 64, k0, 64, 64, 1.f);
    v16bf as, asn;
    load_a16x32_pm(As, row0, 64, k0, 64, as, asn);
    v16bf br0 = load_b32x16(Xr + base, k0, ldb, n0,      H_, 1.f);
    v16bf br1 = load_b32x16(Xr + base, k0, ldb, n0 + 16, H_, 1.f);
    v16bf bi0 = load_b32x16(Xi + base, k0, ldb, n0,      H_, 1.f);
    v16bf bi1 = load_b32x16(Xi + base, k0, ldb, n0 + 16, H_, 1.f);
    gr0 = wmma_bf16(ac,  br0, gr0);  gr0 = wmma_bf16(asn, bi0, gr0);
    gr1 = wmma_bf16(ac,  br1, gr1);  gr1 = wmma_bf16(asn, bi1, gr1);
    gi0 = wmma_bf16(ac,  bi0, gi0);  gi0 = wmma_bf16(as,  br0, gi0);
    gi1 = wmma_bf16(ac,  bi1, gi1);  gi1 = wmma_bf16(as,  br1, gi1);
  }
  store_c(Gr + base, row0, ldb, n0,      H_, gr0);
  store_c(Gr + base, row0, ldb, n0 + 16, H_, gr1);
  store_c(Gi + base, row0, ldb, n0,      H_, gi0);
  store_c(Gi + base, row0, ldb, n0 + 16, H_, gi1);
}

// ---------------- Stage 4/5: block-diag complex MLP (wave tile 16x48) -------
template <int MODE>  // 0: +bias,ReLU   1: +bias,softshrink
__global__ __launch_bounds__(256) void mlp_layer(
    const float* __restrict__ Xr, const float* __restrict__ Xi,
    const float* __restrict__ W0, const float* __restrict__ W1,
    const float* __restrict__ Bi0, const float* __restrict__ Bi1,
    float* __restrict__ Or, float* __restrict__ Oi) {
  const long long total = (N2_ / 16) * 16;     // 8 blocks * 2 half-N groups
  const long long wid = wave_id();
  if (wid >= total) return;
  const long long row0 = (wid / 16) * 16;
  const int r = (int)(wid % 16);
  const int kblk = r >> 1;
  const int n0 = (r & 1) * 48;                 // 0 or 48 within block
  const int colbase = kblk * BS_;
  const float* w0 = W0 + kblk * BS_ * BS_;
  const float* w1 = W1 + kblk * BS_ * BS_;
  v8f orr[3] = {}, oii[3] = {};
#pragma unroll
  for (int k0 = 0; k0 < BS_; k0 += 32) {
    v16bf axr = load_a16x32(Xr + colbase, row0, MED_, k0, N2_, BS_, 1.f);
    v16bf axi = load_a16x32(Xi + colbase, row0, MED_, k0, N2_, BS_, 1.f);
#pragma unroll
    for (int t = 0; t < 3; ++t) {
      const int nn = n0 + 16 * t;
      v16bf bw0 = load_b32x16(w0, k0, BS_, nn, BS_, 1.f);
      v16bf bw1, bw1n;
      load_b32x16_pm(w1, k0, BS_, nn, BS_, bw1, bw1n);
      orr[t] = wmma_bf16(axr, bw0,  orr[t]);
      orr[t] = wmma_bf16(axi, bw1n, orr[t]);
      oii[t] = wmma_bf16(axi, bw0,  oii[t]);
      oii[t] = wmma_bf16(axr, bw1,  oii[t]);
    }
  }
  const int lane = threadIdx.x & 31;
#pragma unroll
  for (int t = 0; t < 3; ++t) {
    const int ncol = n0 + 16 * t + (lane & 15);
    const float br = Bi0[kblk * BS_ + ncol];
    const float bi = Bi1[kblk * BS_ + ncol];
#pragma unroll
    for (int j = 0; j < 8; ++j) {
      float vr = orr[t][j] + br, vi = oii[t][j] + bi;
      if (MODE == 0) {
        vr = vr > 0.f ? vr : 0.f;
        vi = vi > 0.f ? vi : 0.f;
      } else {
        float tr = fabsf(vr) - LAMBDA_;
        float ti = fabsf(vi) - LAMBDA_;
        vr = tr > 0.f ? __builtin_copysignf(tr, vr) : 0.f;
        vi = ti > 0.f ? __builtin_copysignf(ti, vi) : 0.f;
      }
      orr[t][j] = vr; oii[t][j] = vi;
    }
    store_c(Or + colbase, row0, MED_, n0 + 16 * t, N2_, orr[t]);
    store_c(Oi + colbase, row0, MED_, n0 + 16 * t, N2_, oii[t]);
  }
}

// ---------------- Stage 7: inverse real DFT along W + skip (tile 16x32) -----
__global__ __launch_bounds__(256) void inv_wdft_skip(
    const float* __restrict__ Xr, const float* __restrict__ Xi,
    const float* __restrict__ dft, const float* __restrict__ Hskip,
    float* __restrict__ Sp) {
  const long long total = (long long)B_ * H_ * 4 * 24;
  const long long wid = wave_id();
  if (wid >= total) return;
  const int slice = (int)(wid / 96);
  const int r = (int)(wid % 96);
  const int row0 = (r / 24) * 16, n0 = (r % 24) * 32;
  const long long base = (long long)slice * WF_ * MED_;
  v8f a0 = {}, a1 = {};
  {
    v16bf ar = load_a16x32(dft + 20480, row0, 32, 0, 64, 32, 1.f);
    v16bf ai = load_a16x32(dft + 22528, row0, 32, 0, 64, 32, 1.f);
    v16bf br0 = load_b32x16(Xr + base, 0, MED_, n0,      WF_, 1.f);
    v16bf br1 = load_b32x16(Xr + base, 0, MED_, n0 + 16, WF_, 1.f);
    v16bf bi0 = load_b32x16(Xi + base, 0, MED_, n0,      WF_, 1.f);
    v16bf bi1 = load_b32x16(Xi + base, 0, MED_, n0 + 16, WF_, 1.f);
    a0 = wmma_bf16(ar, br0, a0);  a0 = wmma_bf16(ai, bi0, a0);
    a1 = wmma_bf16(ar, br1, a1);  a1 = wmma_bf16(ai, bi1, a1);
  }
  const long long hbase = (long long)slice * W_ * MED_;
  const int lane = threadIdx.x & 31;
  const int mh = (lane >> 4) << 3;
  v8f* accs[2] = {&a0, &a1};
#pragma unroll
  for (int t = 0; t < 2; ++t) {
    const int n = n0 + 16 * t + (lane & 15);
#pragma unroll
    for (int j = 0; j < 8; ++j) {
      const int m = row0 + mh + j;
      if (m < W_) {
        const long long idx = hbase + (long long)m * MED_ + n;
        Sp[idx] = (*accs[t])[j] + Hskip[idx];
      }
    }
  }
}

// ---------------- Stage 8: pwconv2 (wave tile 16x64) ------------------------
__global__ __launch_bounds__(256) void gemm_pw2(
    const float* __restrict__ Sp, const float* __restrict__ Wp,
    float* __restrict__ Out) {
  const long long NTN = CIN_ / 64;             // 6
  const long long wid = wave_id();
  if (wid >= (N1_ / 16) * NTN) return;
  const long long row0 = (wid / NTN) * 16;
  const int n0 = (int)(wid % NTN) * 64;
  v8f acc0 = {}, acc1 = {}, acc2 = {}, acc3 = {};
  for (int k0 = 0; k0 < MED_; k0 += 32) {
    v16bf a = load_a16x32(Sp, row0, MED_, k0, N1_, MED_, 1.f);
    acc0 = wmma_bf16(a, load_b32x16(Wp, k0, CIN_, n0,      MED_, 1.f), acc0);
    acc1 = wmma_bf16(a, load_b32x16(Wp, k0, CIN_, n0 + 16, MED_, 1.f), acc1);
    acc2 = wmma_bf16(a, load_b32x16(Wp, k0, CIN_, n0 + 32, MED_, 1.f), acc2);
    acc3 = wmma_bf16(a, load_b32x16(Wp, k0, CIN_, n0 + 48, MED_, 1.f), acc3);
  }
  store_c(Out, row0, CIN_, n0,      N1_, acc0);
  store_c(Out, row0, CIN_, n0 + 16, N1_, acc1);
  store_c(Out, row0, CIN_, n0 + 32, N1_, acc2);
  store_c(Out, row0, CIN_, n0 + 48, N1_, acc3);
}

// ---------------------------------------------------------------------------
extern "C" void kernel_launch(void* const* d_in, const int* in_sizes, int n_in,
                              void* d_out, int out_size, void* d_ws,
                              size_t ws_size, hipStream_t stream) {
  (void)in_sizes; (void)n_in; (void)out_size; (void)ws_size;
  const float* x      = (const float*)d_in[0];
  const float* w_pw1  = (const float*)d_in[1];
  const float* sscale = (const float*)d_in[2];
  const float* sbias  = (const float*)d_in[3];
  const float* w1     = (const float*)d_in[4];   // [2,8,96,96]
  const float* b1     = (const float*)d_in[5];   // [2,8,96]
  const float* w2     = (const float*)d_in[6];
  const float* b2     = (const float*)d_in[7];
  const float* w_pw2  = (const float*)d_in[8];
  float* out = (float*)d_out;
  float* ws  = (float*)d_ws;

  // Workspace partition (floats). Total ~393.6M floats (~1.58 GB).
  float* dft = ws;                         // 32768 reserved
  float* hb  = dft + 32768;                // N1*MED  (h / skip)
  float* fr  = hb  + N1_ * MED_;           // N2*MED each below
  float* fi  = fr  + N2_ * MED_;
  float* gr  = fi  + N2_ * MED_;
  float* gi  = gr  + N2_ * MED_;
  float* tr  = gi  + N2_ * MED_;
  float* ti  = tr  + N2_ * MED_;
  float* sp  = ti  + N2_ * MED_;           // N1*MED

  const int TPB = 256;                     // 8 waves / block
  init_dft<<<1, TPB, 0, stream>>>(dft);

  // pwconv1 + StarReLU : 6272 * 12 wave-tiles
  gemm_pw1_star<<<9408, TPB, 0, stream>>>(x, w_pw1, sscale, sbias, hb);

  // rfft2 = rDFT along W (TDM-staged A), then complex DFT along H
  fwd_wdft<<<10752, TPB, 0, stream>>>(hb, dft, fr, fi);
  hdft_complex<<<11136, TPB, 0, stream>>>(fr, fi, dft + 4096, dft + 8192,
                                          gr, gi);

  // block-diagonal complex MLP
  mlp_layer<0><<<6496, TPB, 0, stream>>>(gr, gi, w1, w1 + 73728,
                                         b1, b1 + 768, tr, ti);
  mlp_layer<1><<<6496, TPB, 0, stream>>>(tr, ti, w2, w2 + 73728,
                                         b2, b2 + 768, fr, fi);

  // irfft2 = inverse complex DFT along H, inverse rDFT along W (+skip)
  hdft_complex<<<11136, TPB, 0, stream>>>(fr, fi, dft + 12288, dft + 16384,
                                          gr, gi);
  inv_wdft_skip<<<21504, TPB, 0, stream>>>(gr, gi, dft, hb, sp);

  // pwconv2 : 6272 * 6 wave-tiles
  gemm_pw2<<<4704, TPB, 0, stream>>>(sp, w_pw2, out);
}
